// PolynomialExponential_50190987821107
// MI455X (gfx1250) — matrix operand order repaired and checked by reading, not compiled, not run on hardware
//
#include <hip/hip_runtime.h>

// Problem constants (from the reference)
#define NN    8192          // matrix dim
#define FF    128           // feature dim
#define NCOEF 11            // K=10 -> 11 coefficients

// GEMM tiling
#define SPLITK 8
#define KSPLIT (NN / SPLITK)    // 1024 K per split
#define KTILE  64
#define NSTEP  (KSPLIT / KTILE) // 16 K-steps per workgroup
#define MTILE  128
#define LDSW   72               // 64 + 8 pad (bf16 elements) to avoid bank conflicts

// CDNA5 async global->LDS path (ASYNCcnt-tracked DMA), with fallback
#if defined(__has_builtin)
#if __has_builtin(__builtin_amdgcn_global_load_async_to_lds_b128)
#define USE_ASYNC_LDS 1
#endif
#endif
#ifndef USE_ASYNC_LDS
#define USE_ASYNC_LDS 0
#endif

#if defined(__has_builtin)
#if __has_builtin(__builtin_amdgcn_s_wait_asynccnt)
#define WAIT_ASYNC(n) __builtin_amdgcn_s_wait_asynccnt(n)
#endif
#endif
#ifndef WAIT_ASYNC
#define WAIT_ASYNC(n) asm volatile("s_wait_asynccnt %0" ::"i"(n) : "memory")
#endif

// Builtin prototypes use GCC-style generic vectors; pointer types must match.
typedef int v4i_gcc __attribute__((vector_size(16)));
#define ASGLOBAL __attribute__((address_space(1)))
#define ASLDS    __attribute__((address_space(3)))

typedef __bf16 bf16x8  __attribute__((ext_vector_type(8)));
typedef __bf16 bf16x16 __attribute__((ext_vector_type(16)));
typedef float  f32x8   __attribute__((ext_vector_type(8)));

// ---------------------------------------------------------------------------
// 1) Convert L (fp32, 256MB, HBM) -> bf16 (128MB, L2-resident for all GEMMs)
// ---------------------------------------------------------------------------
__global__ __launch_bounds__(256) void cvt_L_kernel(const float* __restrict__ L,
                                                    __bf16* __restrict__ Lbf) {
  size_t base = ((size_t)blockIdx.x * 256 + threadIdx.x) * 8;
  float4 f0 = *(const float4*)(L + base);
  float4 f1 = *(const float4*)(L + base + 4);
  union { __bf16 h[8]; uint4 u; } o;
  o.h[0] = (__bf16)f0.x; o.h[1] = (__bf16)f0.y;
  o.h[2] = (__bf16)f0.z; o.h[3] = (__bf16)f0.w;
  o.h[4] = (__bf16)f1.x; o.h[5] = (__bf16)f1.y;
  o.h[6] = (__bf16)f1.z; o.h[7] = (__bf16)f1.w;
  *(uint4*)(Lbf + base) = o.u;
}

// ---------------------------------------------------------------------------
// 2) prep: y_i = (i==0 ? x : sum of split-K partials);  acc += c_i * y_i;
//    write y_i transposed as bf16 (yT[n][k]) for the next GEMM's B operand.
//    Deterministic reduction (fixed order) instead of fp atomics.
// ---------------------------------------------------------------------------
__global__ __launch_bounds__(256) void prep_kernel(const float* __restrict__ x,
                                                   const float* __restrict__ partials,
                                                   const float* __restrict__ coeffs,
                                                   float* __restrict__ acc,
                                                   __bf16* __restrict__ yT,
                                                   int step, int do_conv) {
  int idx = blockIdx.x * 256 + threadIdx.x;   // < NN*FF = 1M
  int k = idx & (NN - 1);                     // consecutive lanes -> consecutive k
  int n = idx >> 13;                          // NN = 2^13
  float y;
  if (step == 0) {
    y = x[(size_t)k * FF + n];
  } else {
    y = 0.0f;
#pragma unroll
    for (int s = 0; s < SPLITK; ++s)
      y += partials[(size_t)s * NN * FF + (size_t)k * FF + n];
  }
  float c = coeffs[step];
  float a = (step == 0) ? 0.0f : acc[(size_t)k * FF + n];
  acc[(size_t)k * FF + n] = a + c * y;
  if (do_conv)
    yT[(size_t)n * NN + k] = (__bf16)y;       // coalesced transposed store
}

// ---------------------------------------------------------------------------
// 3) GEMM: partial[s] = Lbf[m-block, k-split] @ y   (bf16 WMMA, f32 accum)
//    Block: 256 threads = 8 wave32, C-tile 128x128; wave w owns rows [16w,16w+16)
//    across all 128 columns -> 8 accumulator tiles, 8x A-fragment reuse.
//    Async path: GLOBAL_LOAD_ASYNC_TO_LDS_B128, triple-buffered, 2 tiles in
//    flight, no staging VGPRs, one barrier per K-step.
// ---------------------------------------------------------------------------
#if USE_ASYNC_LDS
#define NBUF 3
#else
#define NBUF 2
#endif

__global__ __launch_bounds__(256) void gemm_bf16_kernel(const __bf16* __restrict__ Lbf,
                                                        const __bf16* __restrict__ yT,
                                                        float* __restrict__ partials) {
  __shared__ __align__(16) __bf16 As[NBUF][MTILE][LDSW];  // A tiles [m][k]
  __shared__ __align__(16) __bf16 Bs[NBUF][FF][LDSW];     // B^T tiles [n][k]

  const int tid  = threadIdx.x;
  const int w    = tid >> 5;
  const int lane = tid & 31;
  const int m0   = blockIdx.x * MTILE;
  const int s    = blockIdx.y;
  const int kbase = s * KSPLIT;

  // WMMA fragment addressing (ISA 16-bit A 16x32 / B 32x16 lane layouts)
  const int mrow = w * 16 + (lane & 15);
  const int klo  = (lane >> 4) * 8;    // A: lane half selects k-subgroup
  const int kbh  = (lane >> 4) * 16;   // unused: B loads use klo, same k order as A
  const int ncol = lane & 15;

  const int row = tid >> 3;            // staging: 1024 16B chunks, 4/thread
  const int c   = tid & 7;

  f32x8 acc[8];
#pragma unroll
  for (int i = 0; i < 8; ++i) acc[i] = f32x8{};

#if USE_ASYNC_LDS
  // 8 async-DMA instructions per wave per stage (in-order completion)
  auto stage = [&](int buf, int kk) {
#pragma unroll
    for (int t = 0; t < 4; ++t) {
      int q  = tid + t * 256;
      int r2 = q >> 3;
      int c2 = q & 7;
      __builtin_amdgcn_global_load_async_to_lds_b128(
          (ASGLOBAL v4i_gcc*)&Lbf[(size_t)(m0 + r2) * NN + kk + c2 * 8],
          (ASLDS v4i_gcc*)&As[buf][r2][c2 * 8], 0, 0);
      __builtin_amdgcn_global_load_async_to_lds_b128(
          (ASGLOBAL v4i_gcc*)&yT[(size_t)r2 * NN + kk + c2 * 8],
          (ASLDS v4i_gcc*)&Bs[buf][r2][c2 * 8], 0, 0);
    }
  };

  stage(0, kbase);           // tiles 0 and 1 in flight
  stage(1, kbase + KTILE);
#else
  // Fallback: double-buffered global->VGPR->LDS staging (spill-free)
  auto stage = [&](int buf, int kk) {
#pragma unroll
    for (int t = 0; t < 4; ++t) {
      int q  = tid + t * 256;
      int r2 = q >> 3;
      int c2 = q & 7;
      *(uint4*)&As[buf][r2][c2 * 8] =
          *(const uint4*)&Lbf[(size_t)(m0 + r2) * NN + kk + c2 * 8];
      *(uint4*)&Bs[buf][r2][c2 * 8] =
          *(const uint4*)&yT[(size_t)r2 * NN + kk + c2 * 8];
    }
  };

  stage(0, kbase);
#endif

  for (int ks = 0; ks < NSTEP; ++ks) {
#if USE_ASYNC_LDS
    const int cur = ks % NBUF;
    // Oldest stage (tile ks) drained: <= 8 ops left means only tile ks+1's
    // 8 DMA ops may still be in flight (in-order completion).
    if (ks + 1 < NSTEP) {
      WAIT_ASYNC(8);
    } else {
      WAIT_ASYNC(0);
    }
    __syncthreads();  // all waves' tile-ks DMA done; buffer (ks+2)%3 idle
    if (ks + 2 < NSTEP) {
      stage((ks + 2) % NBUF, kbase + (ks + 2) * KTILE);
    }
#else
    const int cur = ks & 1;
    __syncthreads();  // buffer `cur` fully staged; buffer !cur free for reuse
    if (ks + 1 < NSTEP) {
      stage(cur ^ 1, kbase + (ks + 1) * KTILE);  // overlaps with compute below
    }
    if (ks + 2 < NSTEP) {
      // L2 prefetch two tiles ahead (global_prefetch_b8)
      __builtin_prefetch(
          &Lbf[(size_t)(m0 + row) * NN + kbase + (ks + 2) * KTILE + c * 8], 0, 1);
      __builtin_prefetch(
          &yT[(size_t)row * NN + kbase + (ks + 2) * KTILE + c * 8], 0, 1);
    }
#endif

    // Compute on buffer `cur`
#pragma unroll
    for (int kc = 0; kc < 2; ++kc) {
      // A fragment: 16x32 bf16; lane<16 -> K {0..7,16..23}, lane>=16 -> {8..15,24..31}
      bf16x8 a0 = *(const bf16x8*)&As[cur][mrow][kc * 32 + klo];
      bf16x8 a1 = *(const bf16x8*)&As[cur][mrow][kc * 32 + klo + 16];
      bf16x16 a = __builtin_shufflevector(a0, a1, 0, 1, 2, 3, 4, 5, 6, 7,
                                          8, 9, 10, 11, 12, 13, 14, 15);
#pragma unroll
      for (int nb = 0; nb < 8; ++nb) {
        // B fragment: 32x16 bf16; lane n = lane&15, K same as A: {0..7,16..23} / {8..15,24..31}
        bf16x8 b0 = *(const bf16x8*)&Bs[cur][nb * 16 + ncol][kc * 32 + klo];
        bf16x8 b1 = *(const bf16x8*)&Bs[cur][nb * 16 + ncol][kc * 32 + klo + 16];
        bf16x16 b = __builtin_shufflevector(b0, b1, 0, 1, 2, 3, 4, 5, 6, 7,
                                            8, 9, 10, 11, 12, 13, 14, 15);
        acc[nb] = __builtin_amdgcn_wmma_f32_16x16x32_bf16(
            false, a, false, b, (short)0, acc[nb], false, false);
      }
    }

#if !USE_ASYNC_LDS
    __syncthreads();  // readers done before next iteration restages
#endif
  }

  // Writeout: C/D layout -> VGPR r, lane L: m = r + 8*(L>=16), n = L&15
  float* out = partials + (size_t)s * NN * FF;
  const int mw = m0 + w * 16 + ((lane >> 4) * 8);
#pragma unroll
  for (int nb = 0; nb < 8; ++nb) {
#pragma unroll
    for (int r = 0; r < 8; ++r) {
      out[(size_t)(mw + r) * FF + nb * 16 + ncol] = acc[nb][r];
    }
  }
}

// ---------------------------------------------------------------------------
// Host launcher
// ---------------------------------------------------------------------------
extern "C" void kernel_launch(void* const* d_in, const int* in_sizes, int n_in,
                              void* d_out, int out_size, void* d_ws, size_t ws_size,
                              hipStream_t stream) {
  const float* L      = (const float*)d_in[0];
  const float* x      = (const float*)d_in[1];
  const float* coeffs = (const float*)d_in[2];
  float* acc = (float*)d_out;

  // Workspace layout: [ L_bf16 : 128MB ][ yT_bf16 : 2MB ][ partials : 32MB ]
  char* ws = (char*)d_ws;
  __bf16* Lbf = (__bf16*)ws;
  __bf16* yT  = (__bf16*)(ws + (size_t)NN * NN * sizeof(__bf16));
  float* partials =
      (float*)(ws + (size_t)NN * NN * sizeof(__bf16) + (size_t)FF * NN * sizeof(__bf16));

  // 1) L -> bf16 once (the only unavoidable full HBM pass)
  unsigned cvt_blocks = (unsigned)(((size_t)NN * NN) / (8 * 256));
  cvt_L_kernel<<<cvt_blocks, 256, 0, stream>>>(L, Lbf);

  // 2) 11 coefficients: axpy+reduce+convert, then bf16 WMMA GEMM (10x)
  unsigned prep_blocks = (NN * FF) / 256;
  dim3 gemm_grid(NN / MTILE, SPLITK);
  for (int i = 0; i < NCOEF; ++i) {
    prep_kernel<<<prep_blocks, 256, 0, stream>>>(x, partials, coeffs, acc, yT, i,
                                                 (i < NCOEF - 1) ? 1 : 0);
    if (i < NCOEF - 1) {
      gemm_bf16_kernel<<<gemm_grid, 256, 0, stream>>>(Lbf, yT, partials);
    }
  }
}
